// MultiheadSelfAttentionEdgeBias_31928786879007
// MI455X (gfx1250) — matrix operand order repaired and checked by reading, not compiled, not run on hardware
//
#include <hip/hip_runtime.h>
#include <hip/hip_bf16.h>

typedef __attribute__((ext_vector_type(16))) __bf16 v16bf;
typedef __attribute__((ext_vector_type(2)))  __bf16 v2bf;
typedef __attribute__((ext_vector_type(8)))  float  v8f;

#define T_SEQ 1024
#define BATCH 4
#define EMBED 1024
#define HEADS 16
#define DHEAD 64
#define N_QKV (3 * EMBED)
#define M_ROWS (T_SEQ * BATCH)
#define SCALE_Q 0.125f
#define NEG_BIG (-3.0e38f)

// ---------- helpers ----------
// Convert float4 -> 4 packed bf16 (2 x v_cvt_pk_bf16_f32, no repack ops)
static __device__ inline uint2 pack4(float4 f) {
    union { v2bf h; unsigned int u; } a, b;
    a.h.x = (__bf16)f.x; a.h.y = (__bf16)f.y;
    b.h.x = (__bf16)f.z; b.h.y = (__bf16)f.w;
    uint2 r; r.x = a.u; r.y = b.u;
    return r;
}

// Build a 16-element bf16 fragment from two 16-byte chunks (p0: elems 0-7, p1: elems 8-15)
static __device__ inline v16bf load_frag(const __bf16* p0, const __bf16* p1) {
    union { v16bf v; uint4 u[2]; } r;
    r.u[0] = *(const uint4*)p0;
    r.u[1] = *(const uint4*)p1;
    return r.v;
}

static __device__ inline v8f zero8() {
    v8f z;
#pragma unroll
    for (int i = 0; i < 8; ++i) z[i] = 0.0f;
    return z;
}

static __device__ inline v8f wmma_bf16(v16bf a, v16bf b, v8f c) {
    return __builtin_amdgcn_wmma_f32_16x16x32_bf16(false, a, false, b, (short)0, c, false, false);
}

// max/sum across the 16-lane half (lanes 0-15 and 16-31 reduce independently)
static __device__ inline float hmax16(float v) {
#pragma unroll
    for (int off = 1; off < 16; off <<= 1) v = fmaxf(v, __shfl_xor(v, off, 32));
    return v;
}
static __device__ inline float hsum16(float v) {
#pragma unroll
    for (int off = 1; off < 16; off <<= 1) v += __shfl_xor(v, off, 32);
    return v;
}

// ================= Kernel 1: QKV projection GEMM (bf16 WMMA) =================
// C[M=4096, N=3072] = X[M,1024] * Wqkv[N,1024]^T + b ; scatter q(*scale)/k/v as bf16 [B,H,T,Dh]
__global__ __launch_bounds__(128) void gemm_qkv_kernel(
    const float* __restrict__ X, const float* __restrict__ W,
    const float* __restrict__ bqkv,
    __bf16* __restrict__ qbuf, __bf16* __restrict__ kbuf, __bf16* __restrict__ vbuf)
{
    __shared__ __bf16 At[64][40];   // 40*2=80B stride (16B aligned)
    __shared__ __bf16 Bt[64][40];

    const int tid  = threadIdx.x;
    const int lane = tid & 31;
    const int wave = tid >> 5;
    const int wm = (wave >> 1) * 32;
    const int wn = (wave & 1) * 32;
    const int m0 = blockIdx.y * 64;
    const int n0 = blockIdx.x * 64;
    const int lrow = lane & 15;
    const int hi8 = (lane < 16) ? 0 : 8;
    const int kbA = (lane < 16) ? 0 : 8;    // A-matrix K base per lane-half
    const int kbB = (lane < 16) ? 0 : 16;   // unused; B fragments also use kbA

    // q/k/v section and head are UNIFORM per block: tile width 64 divides both
    // the 1024-wide q/k/v sections and the 64-wide heads.
    const int sec = n0 >> 10;                     // 0=q 1=k 2=v
    const int h   = (n0 & 1023) >> 6;             // head, uniform
    __bf16* dstbuf = (sec == 0) ? qbuf : ((sec == 1) ? kbuf : vbuf);
    const float scale = (sec == 0) ? SCALE_Q : 1.0f;

    v8f acc[2][2];
#pragma unroll
    for (int i = 0; i < 2; ++i)
#pragma unroll
        for (int j = 0; j < 2; ++j) acc[i][j] = zero8();

    for (int k0 = 0; k0 < EMBED; k0 += 32) {
        // stage A (x) 64x32 f32 -> bf16
#pragma unroll
        for (int it = 0; it < 4; ++it) {
            int q = tid + 128 * it;          // 0..511
            int r = q >> 3;                  // row 0..63
            int c = (q & 7) * 4;             // col 0..28
            const float* src = X + (size_t)(m0 + r) * EMBED + k0 + c;
            float4 f = *(const float4*)src;
            if (k0 + 32 < EMBED) __builtin_prefetch(src + 32, 0, 1);
            *(uint2*)&At[r][c] = pack4(f);
        }
        // stage B (w_qkv rows = N) 64x32 f32 -> bf16
#pragma unroll
        for (int it = 0; it < 4; ++it) {
            int q = tid + 128 * it;
            int r = q >> 3;
            int c = (q & 7) * 4;
            const float* src = W + (size_t)(n0 + r) * EMBED + k0 + c;
            float4 f = *(const float4*)src;
            if (k0 + 32 < EMBED) __builtin_prefetch(src + 32, 0, 1);
            *(uint2*)&Bt[r][c] = pack4(f);
        }
        __syncthreads();

        v16bf af[2], bw[2];
#pragma unroll
        for (int mi = 0; mi < 2; ++mi) {
            int rl = wm + mi * 16 + lrow;
            af[mi] = load_frag(&At[rl][kbA], &At[rl][kbA + 16]);
        }
#pragma unroll
        for (int ni = 0; ni < 2; ++ni) {
            int nl = wn + ni * 16 + lrow;
            bw[ni] = load_frag(&Bt[nl][kbA], &Bt[nl][kbA + 16]);
        }
#pragma unroll
        for (int mi = 0; mi < 2; ++mi)
#pragma unroll
            for (int ni = 0; ni < 2; ++ni)
                acc[mi][ni] = wmma_bf16(af[mi], bw[ni], acc[mi][ni]);
        __syncthreads();
    }

    // epilogue: bias, optional q-scale, scatter to [B,H,T,Dh] bf16 (uniform dst)
#pragma unroll
    for (int mi = 0; mi < 2; ++mi)
#pragma unroll
        for (int ni = 0; ni < 2; ++ni) {
            int d = wn + ni * 16 + lrow;          // 0..63 within head
            float bias = bqkv[n0 + d];
#pragma unroll
            for (int r = 0; r < 8; ++r) {
                int m = m0 + wm + mi * 16 + r + hi8;
                int t = m >> 2, b = m & 3;
                float v = (acc[mi][ni][r] + bias) * scale;
                size_t idx = (((size_t)(b * HEADS + h) * T_SEQ) + t) * DHEAD + d;
                dstbuf[idx] = (__bf16)v;
            }
        }
}

// ================= Kernel 2: flash attention with edge bias =================
// grid: (T/64, B*H). block = 128 (4 waves x 16 query rows).
__global__ __launch_bounds__(128) void attn_kernel(
    const __bf16* __restrict__ qbuf, const __bf16* __restrict__ kbuf,
    const __bf16* __restrict__ vbuf, const float* __restrict__ ebias,
    const unsigned char* __restrict__ kmask, __bf16* __restrict__ attnbuf)
{
    __shared__ __bf16 Kt[32][72];        // [s][d]   144B stride
    __shared__ __bf16 Vt[64][48];        // [d][s]   96B stride (transposed V)
    __shared__ __bf16 Pt[4][16][48];     // per-wave P tile [row][s]

    const int tid  = threadIdx.x;
    const int lane = tid & 31;
    const int wave = tid >> 5;
    const int bh = blockIdx.y;
    const int b  = bh >> 4, h = bh & 15;
    const int q0 = blockIdx.x * 64 + wave * 16;
    const int lrow = lane & 15;
    const int hi8 = (lane < 16) ? 0 : 8;
    const int kbA = (lane < 16) ? 0 : 8;
    const int kbB = (lane < 16) ? 0 : 16;

    // Q fragments (16 rows x 64 K) held in registers for the whole loop
    const __bf16* Qrow = qbuf + ((size_t)bh * T_SEQ + q0 + lrow) * DHEAD;
    v16bf qf0 = load_frag(Qrow + kbA,      Qrow + kbA + 16);
    v16bf qf1 = load_frag(Qrow + 32 + kbA, Qrow + 32 + kbA + 16);

    v8f o[4];
#pragma unroll
    for (int t = 0; t < 4; ++t) o[t] = zero8();
    float mrow[8], lsum[8];
#pragma unroll
    for (int r = 0; r < 8; ++r) { mrow[r] = -INFINITY; lsum[r] = 0.0f; }

    for (int s0 = 0; s0 < T_SEQ; s0 += 32) {
        // stage K tile 32x64 bf16
#pragma unroll
        for (int it = 0; it < 2; ++it) {
            int q = tid + 128 * it;      // 0..255
            int r = q >> 3;              // s-row 0..31
            int c = (q & 7) * 8;         // d 0..56
            const __bf16* src = kbuf + ((size_t)bh * T_SEQ + s0 + r) * DHEAD + c;
            if (s0 + 32 < T_SEQ) __builtin_prefetch(src + 32 * DHEAD, 0, 1);
            *(uint4*)&Kt[r][c] = *(const uint4*)src;
        }
        // stage V tile transposed: Vt[d][s]
#pragma unroll
        for (int it = 0; it < 2; ++it) {
            int q = tid + 128 * it;
            int r = q >> 3;
            int c = (q & 7) * 8;
            const __bf16* src = vbuf + ((size_t)bh * T_SEQ + s0 + r) * DHEAD + c;
            if (s0 + 32 < T_SEQ) __builtin_prefetch(src + 32 * DHEAD, 0, 1);
            uint4 u = *(const uint4*)src;
            const __bf16* e = (const __bf16*)&u;
#pragma unroll
            for (int j = 0; j < 8; ++j) Vt[c + j][r] = e[j];
        }
        __syncthreads();

        // S = Q * K^T  (two 16x16 n-tiles over the 32-key step)
        v8f s[2];
#pragma unroll
        for (int j = 0; j < 2; ++j) {
            int srl = 16 * j + lrow;
            v16bf bk0 = load_frag(&Kt[srl][kbA],      &Kt[srl][kbA + 16]);
            v16bf bk1 = load_frag(&Kt[srl][32 + kbA], &Kt[srl][32 + kbA + 16]);
            v8f c = zero8();
            c = wmma_bf16(qf0, bk0, c);
            c = wmma_bf16(qf1, bk1, c);
            s[j] = c;
        }
        // edge bias + padding mask (dominant HBM stream: 256MB of f32 bias)
#pragma unroll
        for (int j = 0; j < 2; ++j) {
            int sg = s0 + 16 * j + lrow;
            bool msk = kmask[b * T_SEQ + sg] != 0;
            const float* bp = ebias + ((size_t)bh * T_SEQ + q0 + hi8) * T_SEQ + sg;
#pragma unroll
            for (int r = 0; r < 8; ++r)
                s[j][r] = msk ? NEG_BIG : (s[j][r] + bp[(size_t)r * T_SEQ]);
        }
        // online softmax per row; write P tile to per-wave LDS
#pragma unroll
        for (int r = 0; r < 8; ++r) {
            float v = fmaxf(s[0][r], s[1][r]);
            v = hmax16(v);
            float mnew = fmaxf(mrow[r], v);
            float alpha = __expf(mrow[r] - mnew);
            mrow[r] = mnew;
            float p0 = __expf(s[0][r] - mnew);
            float p1 = __expf(s[1][r] - mnew);
            float rs = hsum16(p0 + p1);
            lsum[r] = lsum[r] * alpha + rs;
#pragma unroll
            for (int t = 0; t < 4; ++t) o[t][r] *= alpha;
            Pt[wave][r + hi8][lrow]      = (__bf16)p0;
            Pt[wave][r + hi8][16 + lrow] = (__bf16)p1;
        }
        __builtin_amdgcn_wave_barrier();
        asm volatile("s_wait_dscnt 0x0" ::: "memory"); // intra-wave P store->load
        // O += P * V
        v16bf pf = load_frag(&Pt[wave][lrow][kbA], &Pt[wave][lrow][kbA + 16]);
#pragma unroll
        for (int t = 0; t < 4; ++t) {
            int dcol = 16 * t + lrow;
            v16bf bv = load_frag(&Vt[dcol][kbA], &Vt[dcol][kbA + 16]);
            o[t] = wmma_bf16(pf, bv, o[t]);
        }
        __syncthreads();
    }

    // normalize and write [m = t*B+b, col = h*64+d] bf16
#pragma unroll
    for (int t = 0; t < 4; ++t) {
        int col = h * DHEAD + 16 * t + lrow;
#pragma unroll
        for (int r = 0; r < 8; ++r) {
            int trow = q0 + r + hi8;
            attnbuf[((size_t)trow * BATCH + b) * EMBED + col] = (__bf16)(o[t][r] / lsum[r]);
        }
    }
}

// ================= Kernel 3: output projection GEMM =================
// out[M=4096, N=1024] f32 = A_bf16[M,1024] * Wproj[N,1024]^T + b_proj
__global__ __launch_bounds__(128) void gemm_proj_kernel(
    const __bf16* __restrict__ A, const float* __restrict__ W,
    const float* __restrict__ bp, float* __restrict__ out)
{
    __shared__ __bf16 At[64][40];
    __shared__ __bf16 Bt[64][40];

    const int tid  = threadIdx.x;
    const int lane = tid & 31;
    const int wave = tid >> 5;
    const int wm = (wave >> 1) * 32;
    const int wn = (wave & 1) * 32;
    const int m0 = blockIdx.y * 64;
    const int n0 = blockIdx.x * 64;
    const int lrow = lane & 15;
    const int hi8 = (lane < 16) ? 0 : 8;
    const int kbA = (lane < 16) ? 0 : 8;
    const int kbB = (lane < 16) ? 0 : 16;

    v8f acc[2][2];
#pragma unroll
    for (int i = 0; i < 2; ++i)
#pragma unroll
        for (int j = 0; j < 2; ++j) acc[i][j] = zero8();

    for (int k0 = 0; k0 < EMBED; k0 += 32) {
        // stage A (already bf16): 64x32 = 256 x uint4
#pragma unroll
        for (int it = 0; it < 2; ++it) {
            int q = tid + 128 * it;      // 0..255
            int r = q >> 2;              // row 0..63
            int c = (q & 3) * 8;         // col 0..24
            const __bf16* src = A + (size_t)(m0 + r) * EMBED + k0 + c;
            if (k0 + 32 < EMBED) __builtin_prefetch(src + 32, 0, 1);
            *(uint4*)&At[r][c] = *(const uint4*)src;
        }
        // stage W (f32 -> bf16)
#pragma unroll
        for (int it = 0; it < 4; ++it) {
            int q = tid + 128 * it;
            int r = q >> 3;
            int c = (q & 7) * 4;
            const float* src = W + (size_t)(n0 + r) * EMBED + k0 + c;
            float4 f = *(const float4*)src;
            if (k0 + 32 < EMBED) __builtin_prefetch(src + 32, 0, 1);
            *(uint2*)&Bt[r][c] = pack4(f);
        }
        __syncthreads();

        v16bf af[2], bw[2];
#pragma unroll
        for (int mi = 0; mi < 2; ++mi) {
            int rl = wm + mi * 16 + lrow;
            af[mi] = load_frag(&At[rl][kbA], &At[rl][kbA + 16]);
        }
#pragma unroll
        for (int ni = 0; ni < 2; ++ni) {
            int nl = wn + ni * 16 + lrow;
            bw[ni] = load_frag(&Bt[nl][kbA], &Bt[nl][kbA + 16]);
        }
#pragma unroll
        for (int mi = 0; mi < 2; ++mi)
#pragma unroll
            for (int ni = 0; ni < 2; ++ni)
                acc[mi][ni] = wmma_bf16(af[mi], bw[ni], acc[mi][ni]);
        __syncthreads();
    }

#pragma unroll
    for (int mi = 0; mi < 2; ++mi)
#pragma unroll
        for (int ni = 0; ni < 2; ++ni) {
            int n = n0 + wn + ni * 16 + lrow;
            float bias = bp[n];
#pragma unroll
            for (int r = 0; r < 8; ++r) {
                int m = m0 + wm + mi * 16 + r + hi8;
                out[(size_t)m * EMBED + n] = acc[mi][ni][r] + bias;
            }
        }
}

// ================= launcher =================
extern "C" void kernel_launch(void* const* d_in, const int* in_sizes, int n_in,
                              void* d_out, int out_size, void* d_ws, size_t ws_size,
                              hipStream_t stream) {
    const float*         x      = (const float*)d_in[0];
    const unsigned char* kmask  = (const unsigned char*)d_in[1];
    const float*         ebias  = (const float*)d_in[2];
    const float*         w_qkv  = (const float*)d_in[3];
    const float*         b_qkv  = (const float*)d_in[4];
    const float*         w_proj = (const float*)d_in[5];
    const float*         b_proj = (const float*)d_in[6];
    float*               out    = (float*)d_out;

    const size_t HBUF = (size_t)BATCH * HEADS * T_SEQ * DHEAD; // 4.19M elems
    __bf16* qbuf    = (__bf16*)d_ws;
    __bf16* kbuf    = qbuf + HBUF;
    __bf16* vbuf    = kbuf + HBUF;
    __bf16* attnbuf = vbuf + HBUF;                             // M_ROWS*EMBED elems

    dim3 blk(128, 1, 1);
    gemm_qkv_kernel<<<dim3(N_QKV / 64, M_ROWS / 64), blk, 0, stream>>>(
        x, w_qkv, b_qkv, qbuf, kbuf, vbuf);
    attn_kernel<<<dim3(T_SEQ / 64, BATCH * HEADS), blk, 0, stream>>>(
        qbuf, kbuf, vbuf, ebias, kmask, attnbuf);
    gemm_proj_kernel<<<dim3(EMBED / 64, M_ROWS / 64), blk, 0, stream>>>(
        attnbuf, w_proj, b_proj, out);
}
